// CustomAttention_69389491634476
// MI455X (gfx1250) — matrix operand order repaired and checked by reading, not compiled, not run on hardware
//
#include <hip/hip_runtime.h>
#include <hip/hip_bf16.h>

// ---- problem constants (from reference) ----
#define BB   4
#define NN   2048
#define DD   512
#define HH   8
#define DVV  64
#define DOO  512

typedef __attribute__((ext_vector_type(16))) __bf16 v16bf;
typedef __attribute__((ext_vector_type(8)))  __bf16 v8bf;
typedef __attribute__((ext_vector_type(8)))  float  v8f;

union AFrag { v16bf v; v8bf h[2]; };

// A-fragment (16x32 bf16, ISA 7.12.2 16-bit A layout):
// lane L: row M = L&15 ; group g = L>>4 ; K values = {g*8..g*8+7, 16+g*8..16+g*8+7}
__device__ inline v16bf load_a_frag(const __bf16* A, int lda, int row0, int k0) {
    int lane = threadIdx.x & 31;
    int g = lane >> 4;
    const __bf16* p = A + (size_t)(row0 + (lane & 15)) * lda + k0 + g * 8;
    AFrag f;
    f.h[0] = *(const v8bf*)(p);
    f.h[1] = *(const v8bf*)(p + 16);
    return f.v;
}

// B-fragment (32x16 bf16). Bt stored [n][k] row-major; lane L: col N = L&15,
// K = {g*8..g*8+7, 16+g*8..16+g*8+7}, g = L>>4 (same as A-fragment).
__device__ inline v16bf load_b_frag(const __bf16* Bt, int ldb, int col0, int k0) {
    int lane = threadIdx.x & 31;
    int g = lane >> 4;
    const __bf16* p = Bt + (size_t)(col0 + (lane & 15)) * ldb + k0 + g * 8;
    AFrag f;
    f.h[0] = *(const v8bf*)(p);
    f.h[1] = *(const v8bf*)(p + 16);
    return f.v;
}

__device__ inline v8f wmma_bf16(v16bf a, v16bf b, v8f c) {
    return __builtin_amdgcn_wmma_f32_16x16x32_bf16(false, a, false, b, (short)0, c,
                                                   false, false);
}

// CDNA5 async global->LDS copy (ASYNCcnt-tracked, VGPR-free data path).
// LDS byte offset is the low 32 bits of the generic shared-memory address.
__device__ inline void async_copy_b128(void* lds, const void* gaddr) {
    asm volatile("global_load_async_to_lds_b128 %0, %1, off"
                 :: "v"((unsigned)(unsigned long long)(uintptr_t)lds),
                    "v"((unsigned long long)(uintptr_t)gaddr)
                 : "memory");
}
#define WAIT_ASYNC(n) asm volatile("s_wait_asynccnt %0" :: "i"(n) : "memory")

// ---------------- conversion kernels ----------------
__global__ void cvt_f32_bf16_kernel(const float* __restrict__ src,
                                    __bf16* __restrict__ dst, int n) {
    int i = blockIdx.x * blockDim.x + threadIdx.x;
    if (i < n) dst[i] = (__bf16)src[i];
}

// dst[z][n][k] = src[z][k][n]  (weight transpose into B-fragment-friendly layout)
__global__ void cvt_transpose_kernel(const float* __restrict__ src,
                                     __bf16* __restrict__ dst, int K, int Nout) {
    int i = blockIdx.x * blockDim.x + threadIdx.x;
    int total = K * Nout;
    if (i >= total) return;
    int z = blockIdx.y;
    int n = i / K, k = i % K;
    dst[(size_t)z * total + i] = (__bf16)src[(size_t)z * total + (size_t)k * Nout + n];
}

// ---------------- generic bf16 WMMA GEMM ----------------
// C[z] = A[z/aZdiv] (MxK row major) * Bt[z % bMod]^T (Bt stored [n][k]).
// Block = 8 waves = 128x64 C tile; B tile (64x32) double-buffered in LDS via
// async copies, shared by all waves; each wave: 1 A frag x 4 B frags per K step.
__global__ __launch_bounds__(256) void gemm_bf16_kernel(
    const __bf16* __restrict__ A, const __bf16* __restrict__ Bt,
    void* __restrict__ Cout,
    int M, int Kd, int Nout,
    int aZdiv, long long aZstride, int bMod, long long bZstride,
    long long cZstride, int ldcT, int f32out) {
    __shared__ __align__(64) __bf16 sB[2][64 * 32];
    int t = threadIdx.x;
    int wave = t >> 5;
    int lane = t & 31;
    int nBlocks = Nout >> 6;
    int mB = blockIdx.x / nBlocks, nB = blockIdx.x % nBlocks;
    int row0 = mB * 128 + wave * 16, col0 = nB * 64;
    int z = blockIdx.y;
    const __bf16* Az = A + (size_t)(z / aZdiv) * aZstride;
    const __bf16* Bz = Bt + (size_t)(z % bMod) * bZstride;

    // one 16B async copy per thread per K-step stages the 64x32 B tile
    auto stageB = [&](int s, int bufi) {
        int f = t * 16;                  // byte index in tile
        int nl = f >> 6;                 // row (64B = 32 bf16 per row)
        int wi = f & 63;
        const char* g = (const char*)Bz + ((size_t)(col0 + nl) * Kd) * 2 +
                        (size_t)s * 64 + wi;
        async_copy_b128((char*)&sB[bufi][0] + f, g);
    };

    const int S = Kd >> 5;
    stageB(0, 0);
    v8f acc0 = {}, acc1 = {}, acc2 = {}, acc3 = {};
    for (int s = 0; s < S; ++s) {
        if (s + 1 < S) { stageB(s + 1, (s + 1) & 1); WAIT_ASYNC(1); }
        else          { WAIT_ASYNC(0); }
        __syncthreads();                  // stage s visible to all waves
        const __bf16* sb = &sB[s & 1][0];
        v16bf a  = load_a_frag(Az, Kd, row0, s * 32);
        v16bf b0 = load_b_frag(sb, 32, 0, 0);
        v16bf b1 = load_b_frag(sb, 32, 16, 0);
        v16bf b2 = load_b_frag(sb, 32, 32, 0);
        v16bf b3 = load_b_frag(sb, 32, 48, 0);
        acc0 = wmma_bf16(a, b0, acc0);
        acc1 = wmma_bf16(a, b1, acc1);
        acc2 = wmma_bf16(a, b2, acc2);
        acc3 = wmma_bf16(a, b3, acc3);
        __syncthreads();                  // safe to overwrite buffer s&1
    }
    int g = lane >> 4, ln = lane & 15;
    v8f accs[4] = {acc0, acc1, acc2, acc3};
    for (int tt = 0; tt < 4; ++tt) {
        for (int r = 0; r < 8; ++r) {
            int row = row0 + r + 8 * g;
            int col = col0 + tt * 16 + ln;
            float v = accs[tt][r];
            size_t idx;
            if (ldcT > 0) idx = (size_t)cZstride * z + (size_t)col * ldcT + row;
            else          idx = (size_t)cZstride * z + (size_t)row * Nout + col;
            if (f32out) ((float*)Cout)[idx] = v;
            else        ((__bf16*)Cout)[idx] = (__bf16)v;
        }
    }
}

// ---------------- fused attention ----------------
// Per wave: 16 query rows; Q panel (16x512) held in registers (16 A-frags).
// Per 32-key step: K chunk (32x512, 32KB) + V tile (64x32, 4KB) double-buffered
// in LDS via async copies shared by 8 waves. S streamed to attn output with
// mask+scale, repacked bf16 via per-wave LDS tile (cross-lane transpose), then
// 4 WMMAs accumulate QKV (V stored transposed [dv][n] == B-fragment layout).
__global__ __launch_bounds__(256) void attn_fused_kernel(
    const __bf16* __restrict__ Qp, const __bf16* __restrict__ Kp,
    const __bf16* __restrict__ Vt, const int* __restrict__ mask,
    float* __restrict__ attnOut, __bf16* __restrict__ QKVbf) {
    __shared__ __align__(64) __bf16 sK[2][32 * 512];   // 2 x 32KB
    __shared__ __align__(64) __bf16 sV[2][64 * 32];    // 2 x 4KB
    __shared__ __align__(64) __bf16 sT[8][16 * 32];    // per-wave transpose tile
    int t = threadIdx.x;
    int wave = t >> 5;
    int lane = t & 31;
    int g = lane >> 4, ln = lane & 15;
    int bh = blockIdx.y;
    int b = bh >> 3, h = bh & 7;
    int row0 = blockIdx.x * 128 + wave * 16;

    const __bf16* Q = Qp + (size_t)bh * NN * DD;
    const __bf16* K = Kp + (size_t)bh * NN * DD;
    const __bf16* V = Vt + (size_t)bh * DVV * NN;
    float* S = attnOut + (size_t)bh * NN * NN;
    const float scale = 0.04419417382415922f;  // 1/sqrt(512)

    // Q panel resident in registers: 16 A-fragments (uses >256-VGPR mode)
    v16bf qf[16];
#pragma unroll
    for (int i = 0; i < 16; ++i) qf[i] = load_a_frag(Q, DD, row0, i * 32);

    // stage K rows [s*32, s*32+32) x all 512 e  (8 x 16B per thread)
    // plus V tile [0..64) x k-chunk            (1 x 16B per thread)
    auto stage = [&](int s, int bufi) {
#pragma unroll
        for (int i = 0; i < 8; ++i) {
            int f = (t + i * 256) * 16;       // byte index in 32KB tile
            int r = f >> 10;                  // key row (1024B per row)
            int wi = f & 1023;
            const char* gsrc = (const char*)K + ((size_t)(s * 32 + r)) * 1024 + wi;
            async_copy_b128((char*)&sK[bufi][0] + f, gsrc);
        }
        int f = t * 16;
        int dv = f >> 6;                      // V row (64B per row)
        int wi = f & 63;
        const char* gsrc = (const char*)V + (size_t)dv * (NN * 2) +
                           (size_t)s * 64 + wi;
        async_copy_b128((char*)&sV[bufi][0] + f, gsrc);
    };

    const int STEPS = NN / 32;                // 64
    stage(0, 0);
    v8f acc0 = {}, acc1 = {}, acc2 = {}, acc3 = {};
    for (int s = 0; s < STEPS; ++s) {
        if (s + 1 < STEPS) { stage(s + 1, (s + 1) & 1); WAIT_ASYNC(9); }
        else              { WAIT_ASYNC(0); }
        __syncthreads();
        const __bf16* kp = &sK[s & 1][0];
        const __bf16* vp = &sV[s & 1][0];
        int mp = s * 32;

        v8f s0 = {}, s1 = {};
#pragma unroll
        for (int e = 0; e < 16; ++e) {
            v16bf bk0 = load_b_frag(kp, 512, 0, e * 32);
            v16bf bk1 = load_b_frag(kp, 512, 16, e * 32);
            s0 = wmma_bf16(qf[e], bk0, s0);
            s1 = wmma_bf16(qf[e], bk1, s1);
        }
        // mask + scale, stream to attn output, pack tile into LDS (M-major)
        for (int r = 0; r < 8; ++r) {
            int row = row0 + r + 8 * g;
            int c0 = mp + ln, c1 = mp + 16 + ln;
            float v0 = s0[r] * scale * (1.0f - (float)mask[(size_t)row * NN + c0]);
            float v1 = s1[r] * scale * (1.0f - (float)mask[(size_t)row * NN + c1]);
            S[(size_t)row * NN + c0] = v0;
            S[(size_t)row * NN + c1] = v1;
            int lr = r + 8 * g;
            sT[wave][lr * 32 + ln]      = (__bf16)v0;
            sT[wave][lr * 32 + 16 + ln] = (__bf16)v1;
        }
        // cross-lane transpose via LDS: reload as A-fragment (16x32)
        v16bf a2 = load_a_frag(&sT[wave][0], 32, 0, 0);
        v16bf b0 = load_b_frag(vp, 32, 0, 0);
        v16bf b1 = load_b_frag(vp, 32, 16, 0);
        v16bf b2 = load_b_frag(vp, 32, 32, 0);
        v16bf b3 = load_b_frag(vp, 32, 48, 0);
        acc0 = wmma_bf16(a2, b0, acc0);
        acc1 = wmma_bf16(a2, b1, acc1);
        acc2 = wmma_bf16(a2, b2, acc2);
        acc3 = wmma_bf16(a2, b3, acc3);
        __syncthreads();                      // buffers s&1 reusable
    }
    // QKV bf16 laid out [b][n][h*DV+dv] so the final GEMM A is contiguous
    v8f accs[4] = {acc0, acc1, acc2, acc3};
    for (int tt = 0; tt < 4; ++tt)
        for (int r = 0; r < 8; ++r) {
            int row = row0 + r + 8 * g;
            int dv = tt * 16 + ln;
            QKVbf[(size_t)(b * NN + row) * (HH * DVV) + h * DVV + dv] =
                (__bf16)accs[tt][r];
        }
}

extern "C" void kernel_launch(void* const* d_in, const int* in_sizes, int n_in,
                              void* d_out, int out_size, void* d_ws, size_t ws_size,
                              hipStream_t stream) {
    const float* query = (const float*)d_in[0];
    const float* key   = (const float*)d_in[1];
    const float* value = (const float*)d_in[2];
    const int*   amask = (const int*)d_in[3];
    const float* Wq    = (const float*)d_in[4];
    const float* Wk    = (const float*)d_in[5];
    const float* Wv    = (const float*)d_in[6];
    const float* Wo    = (const float*)d_in[7];

    float* outMain = (float*)d_out;                               // [B,N,DO]
    float* attnOut = outMain + (size_t)BB * NN * DOO;             // [B,H,N,N]

    char* ws = (char*)d_ws;
    size_t off = 0;
    auto alloc = [&](size_t bytes) -> void* {
        void* p = ws + off;
        off = (off + bytes + 255) & ~(size_t)255;
        return p;
    };
    const size_t nXD = (size_t)BB * NN * DD;
    __bf16* qbf  = (__bf16*)alloc(nXD * 2);
    __bf16* kbf  = (__bf16*)alloc(nXD * 2);
    __bf16* vbf  = (__bf16*)alloc(nXD * 2);
    __bf16* WqT  = (__bf16*)alloc((size_t)HH * DD * DD * 2);
    __bf16* WkT  = (__bf16*)alloc((size_t)HH * DD * DD * 2);
    __bf16* WvT  = (__bf16*)alloc((size_t)HH * DD * DVV * 2);
    __bf16* WoT  = (__bf16*)alloc((size_t)DD * DOO * 2);
    __bf16* Qp   = (__bf16*)alloc((size_t)BB * HH * NN * DD * 2);
    __bf16* Kp   = (__bf16*)alloc((size_t)BB * HH * NN * DD * 2);
    __bf16* Vtp  = (__bf16*)alloc((size_t)BB * HH * DVV * NN * 2);
    __bf16* QKVb = (__bf16*)alloc((size_t)BB * NN * HH * DVV * 2);
    (void)ws_size; (void)in_sizes; (void)n_in; (void)out_size;

    // 1) precision conversion + weight transposes
    {
        int n = (int)nXD;
        int blocks = (n + 255) / 256;
        cvt_f32_bf16_kernel<<<blocks, 256, 0, stream>>>(query, qbf, n);
        cvt_f32_bf16_kernel<<<blocks, 256, 0, stream>>>(key,   kbf, n);
        cvt_f32_bf16_kernel<<<blocks, 256, 0, stream>>>(value, vbf, n);
    }
    cvt_transpose_kernel<<<dim3((DD * DD + 255) / 256, HH), 256, 0, stream>>>(Wq, WqT, DD, DD);
    cvt_transpose_kernel<<<dim3((DD * DD + 255) / 256, HH), 256, 0, stream>>>(Wk, WkT, DD, DD);
    cvt_transpose_kernel<<<dim3((DD * DVV + 255) / 256, HH), 256, 0, stream>>>(Wv, WvT, DD, DVV);
    cvt_transpose_kernel<<<dim3((DD * DOO + 255) / 256, 1), 256, 0, stream>>>(Wo, WoT, DD, DOO);

    // 2) projections  Q = X Wq, K = X Wk  (bf16 out, [bh][n][e])
    {
        dim3 grid((NN / 128) * (DD / 64), BB * HH);   // 128 x 32
        gemm_bf16_kernel<<<grid, 256, 0, stream>>>(
            qbf, WqT, Qp, NN, DD, DD,
            HH, (long long)NN * DD, HH, (long long)DD * DD,
            (long long)NN * DD, 0, 0);
        gemm_bf16_kernel<<<grid, 256, 0, stream>>>(
            kbf, WkT, Kp, NN, DD, DD,
            HH, (long long)NN * DD, HH, (long long)DD * DD,
            (long long)NN * DD, 0, 0);
    }
    // 3) V projection, stored transposed [bh][dv][n]
    {
        dim3 grid((NN / 128) * (DVV / 64), BB * HH);  // 16 x 32
        gemm_bf16_kernel<<<grid, 256, 0, stream>>>(
            vbf, WvT, Vtp, NN, DD, DVV,
            HH, (long long)NN * DD, HH, (long long)DD * DVV,
            (long long)DVV * NN, NN, 0);
    }
    // 4) fused S = QK^T*mask/sqrt(D) -> attnOut, and QKV = S V
    attn_fused_kernel<<<dim3(NN / 128, BB * HH), 256, 0, stream>>>(
        Qp, Kp, Vtp, amask, attnOut, QKVb);
    // 5) out = QKV @ Wo (f32 out)
    {
        dim3 grid(((BB * NN) / 128) * (DOO / 64), 1); // 512 x 1
        gemm_bf16_kernel<<<grid, 256, 0, stream>>>(
            QKVb, WoT, outMain, BB * NN, HH * DVV, DOO,
            1, 0, 1, 0, 0, 0, 1);
    }
}
